// IndRNN_46523085750968
// MI455X (gfx1250) — matrix operand order repaired and checked by reading, not compiled, not run on hardware
//
#include <hip/hip_runtime.h>
#include <hip/hip_bf16.h>

typedef __attribute__((ext_vector_type(16))) __bf16 v16bf;
typedef __attribute__((ext_vector_type(8)))  float  v8f;

#define B_  32
#define T_  4096
#define D_  256
#define H_  256

// ---------------------------------------------------------------------------
// Prep: W [D=256, H=256] f32 row-major  ->  Wt [H, D] bf16 (transposed).
// Puts the B-matrix in K-contiguous-per-column order so WMMA B fragments are
// plain contiguous 32-byte loads.
// ---------------------------------------------------------------------------
__global__ __launch_bounds__(256) void indrnn_prep_Wt(const float* __restrict__ W,
                                                      __bf16* __restrict__ Wt) {
    int idx = blockIdx.x * 256 + threadIdx.x;   // 0 .. 65535
    int k = idx >> 8;                           // row of W (D index)
    int h = idx & 255;                          // col of W (H index)
    Wt[(size_t)h * D_ + k] = (__bf16)W[idx];    // W[idx] == W[k][h]
}

// ---------------------------------------------------------------------------
// Fused IndRNN: per wave, a 16-wide h tile of one batch. For each 16-step
// time chunk: xw tile = x[16xD] @ W[:,16] via 8x v_wmma_f32_16x16x32_bf16
// (bias folded into C), then the sequential relu recurrence is walked through
// the tile in-register with lane shuffles, then the tile is stored.
// ---------------------------------------------------------------------------
__global__ __launch_bounds__(128) void indrnn_fused(const float* __restrict__ x,
                                                    const __bf16* __restrict__ Wt,
                                                    const float* __restrict__ bias,
                                                    const float* __restrict__ u,
                                                    float* __restrict__ out) {
    const int lane = threadIdx.x & 31;
    const int wave = threadIdx.x >> 5;      // 0..3
    const int hq   = blockIdx.x;            // 0..3  (64-wide h slab)
    const int b    = blockIdx.y;            // 0..31
    const int h0   = hq * 64 + wave * 16;   // this wave's h tile
    const int n    = lane & 15;             // column within tile
    const int half = lane >> 4;             // lane half (K/M split)

    // --- resident B fragments: 32x16 bf16 each, K = ks*32 .. ks*32+31 ------
    // Layout: lane n holds column N=n; lanes 0-15 K=0..15, lanes 16-31 K=16..31,
    // VGPR v packs K = kbase+2v, kbase+2v+1  -> 16 contiguous bf16 per lane.
    v16bf bfrag[8];
    {
        const __bf16* wp = Wt + (size_t)(h0 + n) * D_ + half * 16;
        #pragma unroll
        for (int ks = 0; ks < 8; ++ks)
            bfrag[ks] = *(const v16bf*)(wp + ks * 32);
    }

    const float un = u[h0 + n];             // recurrent diagonal weight
    const float bn = bias[h0 + n];          // bias, folded into accumulator
    float hstate = 1.0f;                    // h_0 = ones (replicated per half)

    // per-lane A-row pointer: row M = n, K window base = half*16
    const float* rp = x + ((size_t)b * T_ + n) * (size_t)D_ + half * 16;
    // per-lane output column pointer
    float* obase = out + ((size_t)b * T_) * (size_t)H_ + h0 + n;

    for (int t0 = 0; t0 < T_; t0 += 16) {
        // ---- 16x16 xw tile, C initialized with bias --------------------
        v8f acc;
        #pragma unroll
        for (int r = 0; r < 8; ++r) acc[r] = bn;

        #pragma unroll
        for (int ks = 0; ks < 8; ++ks) {
            const float* p = rp + ks * 32;
            // A fragment 16x32 bf16: lanes 0-15 take K 0..15,
            // lanes 16-31 take K 16..31 (kbase = half*16 in rp).
            float4 f0 = *(const float4*)(p);
            float4 f1 = *(const float4*)(p + 4);
            float4 f2 = *(const float4*)(p + 8);
            float4 f3 = *(const float4*)(p + 12);
            v16bf a;
            a[0]  = (__bf16)f0.x; a[1]  = (__bf16)f0.y;
            a[2]  = (__bf16)f0.z; a[3]  = (__bf16)f0.w;
            a[4]  = (__bf16)f1.x; a[5]  = (__bf16)f1.y;
            a[6]  = (__bf16)f1.z; a[7]  = (__bf16)f1.w;
            a[8]  = (__bf16)f2.x; a[9]  = (__bf16)f2.y;
            a[10] = (__bf16)f2.z; a[11] = (__bf16)f2.w;
            a[12] = (__bf16)f3.x; a[13] = (__bf16)f3.y;
            a[14] = (__bf16)f3.z; a[15] = (__bf16)f3.w;

            acc = __builtin_amdgcn_wmma_f32_16x16x32_bf16(
                /*neg_a=*/false, a, /*neg_b=*/false, bfrag[ks],
                /*c_mod=*/(short)0, acc, /*reuse_a=*/false, /*reuse_b=*/false);
        }
        rp += 16 * D_;
        __builtin_prefetch(rp, 0, 1);   // global_prefetch_b8: next x chunk

        // ---- sequential relu recurrence across the 16 time steps -------
        // C/D layout: VGPR r holds t=t0+r (lanes 0-15) and t=t0+r+8 (16-31).
        // hstate is replicated in both halves; shuffle brings each step's
        // value to every lane; predicated write-back keeps tile layout.
        #pragma unroll
        for (int m = 0; m < 16; ++m) {
            int src  = ((m >> 3) << 4) + n;                  // owning lane
            float v  = __shfl(acc[m & 7], src, 32);          // xw(t0+m, n)+bias
            hstate   = fmaxf(fmaf(un, hstate, v), 0.0f);     // relu(v + u*h)
            acc[m & 7] = (half == (m >> 3)) ? hstate : acc[m & 7];
        }

        // ---- store tile (lane -> contiguous 64B per half-row) ----------
        #pragma unroll
        for (int r = 0; r < 8; ++r)
            obase[(size_t)(t0 + r + 8 * half) * H_] = acc[r];
    }
}

extern "C" void kernel_launch(void* const* d_in, const int* in_sizes, int n_in,
                              void* d_out, int out_size, void* d_ws, size_t ws_size,
                              hipStream_t stream) {
    const float* x    = (const float*)d_in[0];   // [B, T, D] f32
    const float* W    = (const float*)d_in[1];   // [D, H]    f32
    const float* bias = (const float*)d_in[2];   // [H]       f32
    const float* u    = (const float*)d_in[3];   // [H]       f32
    float*       out  = (float*)d_out;           // [B, T, H] f32

    __bf16* Wt = (__bf16*)d_ws;                  // 256*256*2 = 128 KB scratch

    indrnn_prep_Wt<<<dim3(256), dim3(256), 0, stream>>>(W, Wt);

    dim3 grid(4, B_);        // 4 h-slabs x 32 batches = 128 blocks
    dim3 block(128);         // 4 waves, one 16-wide h tile each
    indrnn_fused<<<grid, block, 0, stream>>>(x, Wt, bias, u, out);
}
